// MoE_56822417326284
// MI455X (gfx1250) — compile-verified, matrix operand order checked by reading, not run on hardware
//
#include <hip/hip_runtime.h>

#define Bn 8192
#define Hn 1024
#define En 8
#define TM 128
#define TN 128
#define KB 32
#define LDT 40   // LDS row stride in bf16 elems (32 + 8 pad; 80B = 5*16B, b128-aligned)

typedef __attribute__((ext_vector_type(16))) __bf16 v16bf;
typedef __attribute__((ext_vector_type(8)))  float  v8f;

struct U32x8 { uint4 a, b; };

__device__ __forceinline__ unsigned short f2bf(float f) {
  unsigned u = __builtin_bit_cast(unsigned, f);
  u += 0x7FFFu + ((u >> 16) & 1u);           // round-to-nearest-even
  return (unsigned short)(u >> 16);
}
__device__ __forceinline__ float bf2f(unsigned short h) {
  unsigned u = ((unsigned)h) << 16;
  return __builtin_bit_cast(float, u);
}

// ---------------- zero accumulators ----------------
__global__ void zero_kernel(float* logits2, int* counts, float* S, float* misc) {
  int i = blockIdx.x * blockDim.x + threadIdx.x;
  if (i < Bn * 2) logits2[i] = 0.f;
  if (i < En) { counts[i] = 0; S[i] = 0.f; }
  if (i < 2) misc[i] = 0.f;                  // misc[0]=zsq, misc[1]=ce
}

// ---------------- split x -> bf16 hi/lo ----------------
__global__ void split_x(const float* __restrict__ x,
                        unsigned short* __restrict__ xh,
                        unsigned short* __restrict__ xl) {
  size_t i = (size_t)blockIdx.x * blockDim.x + threadIdx.x;
  float v = x[i];
  unsigned short h = f2bf(v);
  float hf = bf2f(h);
  xh[i] = h;
  xl[i] = f2bf(v - hf);                      // exact residual, then RNE
}

// ------- transpose + split w1[e][k][n] -> w1t[e][n][k] bf16 hi/lo -------
__global__ void split_w1t(const float* __restrict__ w1,
                          unsigned short* __restrict__ wth,
                          unsigned short* __restrict__ wtl) {
  __shared__ float tile[32][33];
  int e = blockIdx.z, kb = blockIdx.x * 32, nb = blockIdx.y * 32;
  int tx = threadIdx.x & 31, ty = threadIdx.x >> 5;      // 32x8
  const float* src = w1 + (size_t)e * Hn * Hn;
#pragma unroll
  for (int s = 0; s < 32; s += 8)
    tile[ty + s][tx] = src[(size_t)(kb + ty + s) * Hn + (nb + tx)];
  __syncthreads();
#pragma unroll
  for (int s = 0; s < 32; s += 8) {
    int n = nb + ty + s, k = kb + tx;
    float v = tile[tx][ty + s];
    unsigned short h = f2bf(v);
    size_t o = ((size_t)e * Hn + n) * Hn + k;
    wth[o] = h;
    wtl[o] = f2bf(v - bf2f(h));
  }
}

// ---------------- router: logits, softmax, argmax, compaction ----------------
__global__ __launch_bounds__(1024) void router_kernel(
    const float* __restrict__ x, const float* __restrict__ gate_w,
    int* __restrict__ idxArr, int* __restrict__ perm, int* __restrict__ counts,
    float* __restrict__ S, float* __restrict__ misc) {
  __shared__ float sgw[En * Hn];             // 32 KB
  __shared__ float sProb[32][En];
  __shared__ float sLse2[32];
  __shared__ int sIdx[32], sSlot[32], sBase[En];
  int tid = threadIdx.x, lane = tid & 31, wid = tid >> 5;
  for (int i = tid; i < En * Hn; i += 1024) sgw[i] = gate_w[i];
  __syncthreads();

  int t = blockIdx.x * 32 + wid;
  const float* xr = x + (size_t)t * Hn;
  float acc[En] = {0.f, 0.f, 0.f, 0.f, 0.f, 0.f, 0.f, 0.f};
  for (int i = lane * 4; i < Hn; i += 128) {
    float4 xv = *(const float4*)(xr + i);
#pragma unroll
    for (int e = 0; e < En; e++) {
      float4 g = *(const float4*)(&sgw[e * Hn + i]);
      acc[e] += xv.x * g.x + xv.y * g.y + xv.z * g.z + xv.w * g.w;
    }
  }
#pragma unroll
  for (int off = 16; off > 0; off >>= 1)
#pragma unroll
    for (int e = 0; e < En; e++) acc[e] += __shfl_xor(acc[e], off, 32);

  if (lane == 0) {
    float mx = acc[0]; int im = 0;
#pragma unroll
    for (int e = 1; e < En; e++) if (acc[e] > mx) { mx = acc[e]; im = e; }
    float se = 0.f, ex[En];
#pragma unroll
    for (int e = 0; e < En; e++) { ex[e] = __expf(acc[e] - mx); se += ex[e]; }
    float lse = mx + logf(se);
#pragma unroll
    for (int e = 0; e < En; e++) sProb[wid][e] = ex[e] / se;
    sIdx[wid] = im;
    sLse2[wid] = lse * lse;
    idxArr[t] = im;
  }
  __syncthreads();
  if (tid == 0) {
    int c[En] = {0, 0, 0, 0, 0, 0, 0, 0};
    for (int w = 0; w < 32; w++) sSlot[w] = c[sIdx[w]]++;
#pragma unroll
    for (int e = 0; e < En; e++)
      if (c[e] > 0) sBase[e] = atomicAdd(&counts[e], c[e]);
    for (int w = 0; w < 32; w++) sSlot[w] += sBase[sIdx[w]];
    float z = 0.f;
    for (int w = 0; w < 32; w++) z += sLse2[w];
    atomicAdd(&misc[0], z);
#pragma unroll
    for (int e = 0; e < En; e++) {
      float s = 0.f;
      for (int w = 0; w < 32; w++) s += sProb[w][e];
      atomicAdd(&S[e], s);
    }
  }
  __syncthreads();
  if (lane == 0) perm[sIdx[wid] * Bn + sSlot[wid]] = blockIdx.x * 32 + wid;
}

// ---------------- grouped GEMM: split-bf16 WMMA + fused GELU/w2 epilogue ----------------
__device__ __forceinline__ v16bf ld_frag(const unsigned short* p) {
  U32x8 u;
  u.a = *(const uint4*)(p);
  u.b = *(const uint4*)(p + 16);
  return __builtin_bit_cast(v16bf, u);
}

__global__ __launch_bounds__(256) void moe_gemm(
    const unsigned short* __restrict__ xh, const unsigned short* __restrict__ xl,
    const unsigned short* __restrict__ wth, const unsigned short* __restrict__ wtl,
    const float* __restrict__ b1, const float* __restrict__ w2,
    const int* __restrict__ perm, const int* __restrict__ counts,
    float* __restrict__ logits2) {
  __shared__ __align__(16) unsigned short sA0[TM * LDT], sA1[TM * LDT];
  __shared__ __align__(16) unsigned short sB0[TN * LDT], sB1[TN * LDT];
  __shared__ int sTok[TM];

  int e = blockIdx.z;
  int cnt = counts[e];
  int mstart = blockIdx.x * TM;
  if (mstart >= cnt) return;
  int n0 = blockIdx.y * TN;
  int tid = threadIdx.x, lane = tid & 31, wid = tid >> 5;

  if (tid < TM) {
    int m = mstart + tid;
    sTok[tid] = perm[e * Bn + (m < cnt ? m : cnt - 1)];
  }
  __syncthreads();

  v8f acc[4][2] = {};
  const size_t wbase = ((size_t)e * Hn + n0) * Hn;
  const int mbase = (wid >> 2) * 64, nbase = (wid & 3) * 32;
  const int rrow = lane & 15;
  const int klo = (lane & 16) ? 8 : 0;

  for (int kk = 0; kk < Hn; kk += KB) {
#pragma unroll
    for (int it = 0; it < 2; ++it) {                 // 512 16B chunks / 256 threads
      int chunk = tid + it * 256;
      int row = chunk >> 2;
      int ko = (chunk & 3) * 8;
      int t = sTok[row];
      size_t ga = (size_t)t * Hn + kk + ko;
      *(uint4*)(sA0 + row * LDT + ko) = *(const uint4*)(xh + ga);
      *(uint4*)(sA1 + row * LDT + ko) = *(const uint4*)(xl + ga);
      size_t gb = wbase + (size_t)row * Hn + kk + ko;
      *(uint4*)(sB0 + row * LDT + ko) = *(const uint4*)(wth + gb);
      *(uint4*)(sB1 + row * LDT + ko) = *(const uint4*)(wtl + gb);
    }
    __syncthreads();

    v16bf ah[4], al[4], bh[2], bl[2];
#pragma unroll
    for (int i = 0; i < 4; i++) {
      int o = (mbase + i * 16 + rrow) * LDT + klo;
      ah[i] = ld_frag(sA0 + o);
      al[i] = ld_frag(sA1 + o);
    }
#pragma unroll
    for (int j = 0; j < 2; j++) {
      int o = (nbase + j * 16 + rrow) * LDT + klo;
      bh[j] = ld_frag(sB0 + o);
      bl[j] = ld_frag(sB1 + o);
    }
#pragma unroll
    for (int i = 0; i < 4; i++)
#pragma unroll
      for (int j = 0; j < 2; j++) {
        acc[i][j] = __builtin_amdgcn_wmma_f32_16x16x32_bf16(
            false, ah[i], false, bh[j], (short)0, acc[i][j], false, false);
        acc[i][j] = __builtin_amdgcn_wmma_f32_16x16x32_bf16(
            false, ah[i], false, bl[j], (short)0, acc[i][j], false, false);
        acc[i][j] = __builtin_amdgcn_wmma_f32_16x16x32_bf16(
            false, al[i], false, bh[j], (short)0, acc[i][j], false, false);
      }
    __syncthreads();
  }

  // ---- epilogue: exact GELU, h @ w2[e], reduce over n, atomic into logits2 ----
  float b1v[2]; float2 w2v[2];
#pragma unroll
  for (int j = 0; j < 2; j++) {
    int n = n0 + nbase + j * 16 + (lane & 15);
    b1v[j] = b1[e * Hn + n];
    w2v[j] = *(const float2*)(w2 + ((size_t)e * Hn + n) * 2);
  }
#pragma unroll
  for (int i = 0; i < 4; i++) {
#pragma unroll
    for (int r = 0; r < 8; r++) {
      float s0 = 0.f, s1 = 0.f;
#pragma unroll
      for (int j = 0; j < 2; j++) {
        float v = acc[i][j][r] + b1v[j];
        float g = 0.5f * v * (1.f + erff(v * 0.70710678118654752f));
        s0 += g * w2v[j].x;
        s1 += g * w2v[j].y;
      }
#pragma unroll
      for (int off = 8; off > 0; off >>= 1) {        // reduce over 16 n-lanes per half
        s0 += __shfl_xor(s0, off, 32);
        s1 += __shfl_xor(s1, off, 32);
      }
      if ((lane & 15) == 0) {
        int ml = mbase + i * 16 + r + ((lane & 16) ? 8 : 0);
        if (mstart + ml < cnt) {
          int t = sTok[ml];
          atomicAdd(&logits2[t * 2 + 0], s0);
          atomicAdd(&logits2[t * 2 + 1], s1);
        }
      }
    }
  }
}

// ---------------- per-token CE: softmax -> log_softmax -> pick label ----------------
__global__ void finalize_tokens(const float* __restrict__ logits2,
                                const int* __restrict__ idxArr,
                                const int* __restrict__ label,
                                const float* __restrict__ b2,
                                float* __restrict__ ce) {
  __shared__ float red[256];
  int t = blockIdx.x * 256 + threadIdx.x;
  int e = idxArr[t];
  float l0 = logits2[t * 2 + 0] + b2[e * 2 + 0];
  float l1 = logits2[t * 2 + 1] + b2[e * 2 + 1];
  float m = fmaxf(l0, l1);
  float e0 = __expf(l0 - m), e1 = __expf(l1 - m);
  float inv = 1.f / (e0 + e1);
  float p0 = e0 * inv, p1 = e1 * inv;            // expert_out (weight == 1)
  float mm = fmaxf(p0, p1);
  float lse = mm + logf(__expf(p0 - mm) + __expf(p1 - mm));
  float pl = (label[t] == 0) ? p0 : p1;
  red[threadIdx.x] = lse - pl;
  __syncthreads();
  for (int s = 128; s > 0; s >>= 1) {
    if (threadIdx.x < s) red[threadIdx.x] += red[threadIdx.x + s];
    __syncthreads();
  }
  if (threadIdx.x == 0) atomicAdd(ce, red[0]);
}

__global__ void finalize_loss(const float* __restrict__ S,
                              const int* __restrict__ counts,
                              const float* __restrict__ misc,
                              float* __restrict__ out) {
  float bal = 0.f;
#pragma unroll
  for (int e = 0; e < En; e++) bal += S[e] * (float)counts[e];
  bal *= (float)En / ((float)Bn * (float)Bn);
  out[0] = misc[1] / (float)Bn + 0.01f * bal + 0.001f * (misc[0] / (float)Bn);
}

// ---------------- launcher ----------------
extern "C" void kernel_launch(void* const* d_in, const int* in_sizes, int n_in,
                              void* d_out, int out_size, void* d_ws, size_t ws_size,
                              hipStream_t stream) {
  (void)in_sizes; (void)n_in; (void)out_size; (void)ws_size;
  const float* x      = (const float*)d_in[0];
  const int*   label  = (const int*)d_in[1];
  const float* gate_w = (const float*)d_in[2];
  const float* w1     = (const float*)d_in[3];
  const float* b1     = (const float*)d_in[4];
  const float* w2     = (const float*)d_in[5];
  const float* b2     = (const float*)d_in[6];
  float* out = (float*)d_out;

  char* ws = (char*)d_ws;
  size_t off = 0;
  auto take = [&](size_t bytes) -> char* {
    char* p = ws + off;
    off = (off + bytes + 255) & ~(size_t)255;
    return p;
  };
  unsigned short* xh   = (unsigned short*)take((size_t)Bn * Hn * 2);
  unsigned short* xl   = (unsigned short*)take((size_t)Bn * Hn * 2);
  unsigned short* wth  = (unsigned short*)take((size_t)En * Hn * Hn * 2);
  unsigned short* wtl  = (unsigned short*)take((size_t)En * Hn * Hn * 2);
  int*   perm    = (int*)take((size_t)En * Bn * 4);
  int*   idxArr  = (int*)take((size_t)Bn * 4);
  float* logits2 = (float*)take((size_t)Bn * 2 * 4);
  int*   counts  = (int*)take(En * 4);
  float* S       = (float*)take(En * 4);
  float* misc    = (float*)take(2 * 4);        // [0]=zsq sum, [1]=ce sum

  zero_kernel<<<(Bn * 2 + 255) / 256, 256, 0, stream>>>(logits2, counts, S, misc);
  split_x<<<(Bn * Hn) / 256, 256, 0, stream>>>(x, xh, xl);
  dim3 gw(Hn / 32, Hn / 32, En);
  split_w1t<<<gw, 256, 0, stream>>>(w1, wth, wtl);
  router_kernel<<<Bn / 32, 1024, 0, stream>>>(x, gate_w, idxArr, perm, counts, S, misc);
  dim3 gg(Bn / TM, Hn / TN, En);
  moe_gemm<<<gg, 256, 0, stream>>>(xh, xl, wth, wtl, b1, w2, perm, counts, logits2);
  finalize_tokens<<<Bn / 256, 256, 0, stream>>>(logits2, idxArr, label, b2, misc + 1);
  finalize_loss<<<1, 1, 0, stream>>>(S, counts, misc, out);
}
